// TelNet_81810537055274
// MI455X (gfx1250) — matrix operand order repaired and checked by reading, not compiled, not run on hardware
//
#include <hip/hip_runtime.h>
#include <hip/hip_bf16.h>
#include <math.h>

// ---------------------------------------------------------------- types
typedef __bf16 bf16_t;
typedef bf16_t v16bf __attribute__((ext_vector_type(16)));
typedef float  v8f   __attribute__((ext_vector_type(8)));

#define B_  32
#define T_  12
#define HH  64
#define WW  128
#define I_  8
#define D_  512
#define L_  6
#define N_  16
#define V_  9

// async-to-LDS availability (compile-guarded; falls back to ds stores)
#if defined(__has_builtin)
#  if __has_builtin(__builtin_amdgcn_global_load_async_to_lds_b128)
#    define HAVE_ASYNC_LDS 1
#  else
#    define HAVE_ASYNC_LDS 0
#  endif
#else
#  define HAVE_ASYNC_LDS 0
#endif

#if HAVE_ASYNC_LDS
// builtin signature (from hipcc diagnostic): param0 = int4 AS(1)*, param1 = int4 AS(3)*, then 2 imm ints
typedef int v4i_ __attribute__((vector_size(16)));
typedef __attribute__((address_space(1))) v4i_ g_v4i;
typedef __attribute__((address_space(3))) v4i_ l_v4i;

__device__ __forceinline__ void async_copy16(const void* g, void* l) {
  __builtin_amdgcn_global_load_async_to_lds_b128(
      (g_v4i*)(unsigned long long)(uintptr_t)g,
      (l_v4i*)(unsigned int)(uintptr_t)l, 0, 0);
}
__device__ __forceinline__ void async_wait0() {
#if __has_builtin(__builtin_amdgcn_s_wait_asynccnt)
  __builtin_amdgcn_s_wait_asynccnt(0);
#else
  asm volatile("s_wait_asynccnt 0" ::: "memory");
#endif
}
#endif

// ---------------------------------------------------------------- helpers
__device__ __forceinline__ unsigned short f2bf(float f) {
  unsigned int u = __float_as_uint(f);
  u += 0x7FFFu + ((u >> 16) & 1u);            // round-to-nearest-even
  return (unsigned short)(u >> 16);
}
__device__ __forceinline__ float sigmoidf_(float x) { return 1.0f / (1.0f + __expf(-x)); }

__device__ __forceinline__ v16bf as_v16bf(int4 a, int4 b) {
  union { int4 i[2]; v16bf v; } u; u.i[0] = a; u.i[1] = b; return u.v;
}
// A fragment (16x32 bf16): lane holds row M=lane&15; K elems [k0..k0+8) and [k0+16..k0+24), k0=(lane>>4)*8
__device__ __forceinline__ v16bf load_a_frag(const unsigned short* row, int kbase, int lane) {
  int k0 = kbase + ((lane >> 4) << 3);
  return as_v16bf(*(const int4*)(row + k0), *(const int4*)(row + k0 + 16));
}
// B fragment (32x16 bf16), B supplied N-major (row n = column of B): K elems [k0..k0+8) and [k0+16..k0+24), k0=(lane>>4)*8
__device__ __forceinline__ v16bf load_b_frag(const unsigned short* col, int kbase, int lane) {
  int k0 = kbase + ((lane >> 4) << 3);
  return as_v16bf(*(const int4*)(col + k0), *(const int4*)(col + k0 + 16));
}

__device__ float blk_sum(float v, float* s_red) {        // blockDim.x == 256
  int t = threadIdx.x;
  s_red[t] = v; __syncthreads();
  for (int o = 128; o > 0; o >>= 1) { if (t < o) s_red[t] += s_red[t + o]; __syncthreads(); }
  float r = s_red[0]; __syncthreads();
  return r;
}

// ---------------------------------------------------------------- small utility kernels
__global__ void k_convert4(const float* in, unsigned short* out, size_t n4) {
  size_t i = (size_t)blockIdx.x * blockDim.x + threadIdx.x;
  if (i >= n4) return;
  float4 v = ((const float4*)in)[i];
  uint2 o;
  o.x = (unsigned int)f2bf(v.x) | ((unsigned int)f2bf(v.y) << 16);
  o.y = (unsigned int)f2bf(v.z) | ((unsigned int)f2bf(v.w) << 16);
  ((uint2*)out)[i] = o;
}
__global__ void k_zero(float* p, size_t n) {
  size_t i = (size_t)blockIdx.x * blockDim.x + threadIdx.x;
  if (i < n) p[i] = 0.0f;
}
__global__ void k_wcat(const float* wih, const float* whh, unsigned short* out) {
  int i = blockIdx.x * blockDim.x + threadIdx.x;           // 2048*1024
  if (i >= 2048 * 1024) return;
  int nrow = i >> 10, k = i & 1023;
  float v = (k < 512) ? wih[nrow * 512 + k] : whh[nrow * 512 + (k - 512)];
  out[i] = f2bf(v);
}

// ---------------------------------------------------------------- mask ratio pipeline (shared mask m0)
__global__ void k_mask(const float* m0, float* ratio1, float* upd1,
                       float* ratio2, float* upd2, float* r3u3) {
  __shared__ float s_m1[2048];
  __shared__ float s_m2[512];
  int t = threadIdx.x;                                     // blockDim 1024
  for (int i = t; i < 2048; i += 1024) {
    int y = i >> 6, x = i & 63;
    const float* p = m0 + (2 * y) * WW + 2 * x;
    float s = p[0] + p[1] + p[WW] + p[WW + 1];
    float u = fminf(fmaxf(s, 0.f), 1.f);
    ratio1[i] = 4.0f / (s + 1e-8f) * u; upd1[i] = u; s_m1[i] = u;
  }
  __syncthreads();
  if (t < 512) {
    int y = t >> 5, x = t & 31;
    float s = s_m1[(2 * y) * 64 + 2 * x] + s_m1[(2 * y) * 64 + 2 * x + 1]
            + s_m1[(2 * y + 1) * 64 + 2 * x] + s_m1[(2 * y + 1) * 64 + 2 * x + 1];
    float u = fminf(fmaxf(s, 0.f), 1.f);
    ratio2[t] = 4.0f / (s + 1e-8f) * u; upd2[t] = u; s_m2[t] = u;
  }
  __syncthreads();
  for (int o = 256; o > 0; o >>= 1) { if (t < o && t + o < 512) s_m2[t] += s_m2[t + o]; __syncthreads(); }
  if (t == 0) {
    float s = s_m2[0];
    float u = fminf(fmaxf(s, 0.f), 1.f);
    r3u3[0] = 512.0f / (s + 1e-8f) * u;
    r3u3[1] = u;
  }
}

// ---------------------------------------------------------------- static roll embedding: Xs (B,18,D)
__global__ void k_xs(const float* Xstatic, const float* static_w, float* Xs) {
  int i = blockIdx.x * blockDim.x + threadIdx.x;           // 32*18*512
  if (i >= B_ * 18 * D_) return;
  int d = i & 511; int bs = i >> 9; int si = bs % 18; int b = bs / 18;
  int s = si - (T_ - 1);
  float a = 0.f;
  for (int j = 0; j < 12; ++j) {
    int jj = (((j - s) % 12) + 12) % 12;
    a += Xstatic[b * 12 + jj] * static_w[d * 12 + j];
  }
  Xs[i] = a;
}

// ---------------------------------------------------------------- fused pconv1 -> im2col A1 (bf16)
// A1[m][k]: m = n*512 + y2*32 + x2, k = ic*4 + ky*2 + kx  (4 values per thread)
__global__ void k_h1(const float* Xauto, const float* m0, const float* conv1_w, const float* conv1_b,
                     const float* ratio1, const float* upd1, unsigned short* A1) {
  int i = blockIdx.x * 256 + threadIdx.x;                  // 384*128*512 = 25165824
  if (i >= 384 * 128 * 512) return;
  int sp = i & 511;
  int ic = (i >> 9) & 127;
  int n  = i >> 16;
  int y2 = sp >> 5, x2 = sp & 31;
  const float* xb = Xauto + (size_t)n * 8192 + (4 * y2) * WW + 4 * x2;
  const float* mb = m0 + (4 * y2) * WW + 4 * x2;
  float xm[4][4];
#pragma unroll
  for (int r = 0; r < 4; ++r) {
    float4 xv = *(const float4*)(xb + r * WW);
    float4 mv = *(const float4*)(mb + r * WW);
    xm[r][0] = xv.x * mv.x; xm[r][1] = xv.y * mv.y;
    xm[r][2] = xv.z * mv.z; xm[r][3] = xv.w * mv.w;
  }
  const float* wp = conv1_w + ic * 4;
  float w0 = wp[0], w1 = wp[1], w2 = wp[2], w3 = wp[3];
  float bi = conv1_b[ic];
  unsigned short tmp[4];
#pragma unroll
  for (int ky = 0; ky < 2; ++ky)
#pragma unroll
    for (int kx = 0; kx < 2; ++kx) {
      int r0 = 2 * ky, c0 = 2 * kx;
      float raw = xm[r0][c0] * w0 + xm[r0][c0 + 1] * w1
                + xm[r0 + 1][c0] * w2 + xm[r0 + 1][c0 + 1] * w3;
      int sp1 = (2 * y2 + ky) * 64 + (2 * x2 + kx);
      float u = upd1[sp1];
      tmp[ky * 2 + kx] = f2bf((raw * ratio1[sp1] + bi) * u * u);   // h1*m1 (conv2 input)
    }
  uint2 o;
  o.x = (unsigned int)tmp[0] | ((unsigned int)tmp[1] << 16);
  o.y = (unsigned int)tmp[2] | ((unsigned int)tmp[3] << 16);
  size_t m = (size_t)n * 512 + sp;
  *(uint2*)(A1 + m * 512 + ic * 4) = o;
}

// ---------------------------------------------------------------- WMMA GEMM: conv2 (M=196608,N=256,K=512)
// one wave = one M tile x 8 N tiles (64 acc VGPRs, no spills) -> A1 read 2x; W2 (256KB) L2-hot
__global__ void __launch_bounds__(256, 1)
k_gemm_conv2(const unsigned short* A1, const unsigned short* W2b,
             const float* conv2_b, const float* ratio2, const float* upd2,
             unsigned short* A2) {
  int wave = (blockIdx.x * blockDim.x + threadIdx.x) >> 5;
  int lane = threadIdx.x & 31;
  if (wave >= 24576) return;                               // 12288 M tiles * 2 N halves
  int half = wave & 1;
  int mt = wave >> 1;
  const unsigned short* arow = A1 + (size_t)(mt * 16 + (lane & 15)) * 512;
  const unsigned short* bbase = W2b + (size_t)(half * 8 * 16 + (lane & 15)) * 512;
  v8f zer = {0.f,0.f,0.f,0.f,0.f,0.f,0.f,0.f};
  v8f acc[8];
#pragma unroll
  for (int j = 0; j < 8; ++j) acc[j] = zer;
  for (int k = 0; k < 512; k += 32) {
    v16bf a = load_a_frag(arow, k, lane);
#pragma unroll
    for (int j = 0; j < 8; ++j) {
      v16bf b = load_b_frag(bbase + (size_t)j * 16 * 512, k, lane);
      acc[j] = __builtin_amdgcn_wmma_f32_16x16x32_bf16(false, a, false, b, (short)0, acc[j], false, false);
    }
  }
#pragma unroll
  for (int j = 0; j < 8; ++j) {
    int n = (half * 8 + j) * 16 + (lane & 15);
    float bn = conv2_b[n];
#pragma unroll
    for (int r = 0; r < 8; ++r) {
      int m = mt * 16 + r + ((lane >> 4) << 3);
      int nimg = m >> 9, sp = m & 511;
      float u = upd2[sp];
      float val = (acc[j][r] * ratio2[sp] + bn) * u * u;   // h2*m2 (conv3 input)
      A2[(size_t)nimg * 131072 + (size_t)n * 512 + sp] = f2bf(val);
    }
  }
}

// ---------------------------------------------------------------- WMMA GEMM: conv3 split-K (M=384,N=512,K=131072)
// 2M x 4N register blocking; chunk-slowest ordering keeps the 7MB K-slice working set L2-resident
__global__ void __launch_bounds__(256, 1)
k_gemm_conv3(const unsigned short* A2, const unsigned short* W3b, float* rawC) {
  int wave = (blockIdx.x * blockDim.x + threadIdx.x) >> 5;
  int lane = threadIdx.x & 31;
  if (wave >= 3072) return;                                // 32 chunks * 12 Mb * 8 Nb
  int chunk = wave / 96;
  int rem = wave % 96;
  int mb = rem % 12;
  int nb = rem / 12;
  const unsigned short* ar0 = A2 + (size_t)((mb * 2 + 0) * 16 + (lane & 15)) * 131072;
  const unsigned short* ar1 = A2 + (size_t)((mb * 2 + 1) * 16 + (lane & 15)) * 131072;
  const unsigned short* br0 = W3b + (size_t)((nb * 4 + 0) * 16 + (lane & 15)) * 131072;
  const unsigned short* br1 = W3b + (size_t)((nb * 4 + 1) * 16 + (lane & 15)) * 131072;
  const unsigned short* br2 = W3b + (size_t)((nb * 4 + 2) * 16 + (lane & 15)) * 131072;
  const unsigned short* br3 = W3b + (size_t)((nb * 4 + 3) * 16 + (lane & 15)) * 131072;
  v8f zer = {0.f,0.f,0.f,0.f,0.f,0.f,0.f,0.f};
  v8f acc[2][4];
#pragma unroll
  for (int x = 0; x < 2; ++x)
#pragma unroll
    for (int j = 0; j < 4; ++j) acc[x][j] = zer;
  int kb = chunk * 4096;
  for (int k = 0; k < 4096; k += 32) {
    v16bf b0 = load_b_frag(br0, kb + k, lane);
    v16bf b1 = load_b_frag(br1, kb + k, lane);
    v16bf b2 = load_b_frag(br2, kb + k, lane);
    v16bf b3 = load_b_frag(br3, kb + k, lane);
    v16bf a0 = load_a_frag(ar0, kb + k, lane);
    acc[0][0] = __builtin_amdgcn_wmma_f32_16x16x32_bf16(false, a0, false, b0, (short)0, acc[0][0], false, false);
    acc[0][1] = __builtin_amdgcn_wmma_f32_16x16x32_bf16(false, a0, false, b1, (short)0, acc[0][1], false, false);
    acc[0][2] = __builtin_amdgcn_wmma_f32_16x16x32_bf16(false, a0, false, b2, (short)0, acc[0][2], false, false);
    acc[0][3] = __builtin_amdgcn_wmma_f32_16x16x32_bf16(false, a0, false, b3, (short)0, acc[0][3], false, false);
    v16bf a1 = load_a_frag(ar1, kb + k, lane);
    acc[1][0] = __builtin_amdgcn_wmma_f32_16x16x32_bf16(false, a1, false, b0, (short)0, acc[1][0], false, false);
    acc[1][1] = __builtin_amdgcn_wmma_f32_16x16x32_bf16(false, a1, false, b1, (short)0, acc[1][1], false, false);
    acc[1][2] = __builtin_amdgcn_wmma_f32_16x16x32_bf16(false, a1, false, b2, (short)0, acc[1][2], false, false);
    acc[1][3] = __builtin_amdgcn_wmma_f32_16x16x32_bf16(false, a1, false, b3, (short)0, acc[1][3], false, false);
  }
#pragma unroll
  for (int x = 0; x < 2; ++x)
#pragma unroll
    for (int j = 0; j < 4; ++j) {
      int n = (nb * 4 + j) * 16 + (lane & 15);
#pragma unroll
      for (int r = 0; r < 8; ++r) {
        int m = (mb * 2 + x) * 16 + r + ((lane >> 4) << 3);
        atomicAdd(&rawC[(size_t)m * 512 + n], acc[x][j][r]);
      }
    }
}

__global__ void k_finish3(float* raw, const float* b3, const float* r3u3) {
  int i = blockIdx.x * blockDim.x + threadIdx.x;
  if (i >= 384 * 512) return;
  int d = i & 511;
  raw[i] = (raw[i] * r3u3[0] + b3[d]) * r3u3[1];
}

// ---------------------------------------------------------------- varsel (GRN variable selection)
struct VSN {
  const float *rn_gate, *rn_scale, *rn_bias;
  const float *fc1_w, *fc1_b, *fc2_w, *fc2_b;
  const float *glu_w, *glu_b, *grn_s, *grn_b;
  const float *rs_gate, *rs_scale, *rs_bias;
};

__device__ void varsel_core(float* s_xin, float* s_red, float* s_mu, float* s_rs,
                            float* s_w, float* s_h, const VSN& p,
                            const float* oln_s, const float* oln_b,
                            unsigned short* out_bf16, float* wgt_out) {
  const int t = threadIdx.x;
  for (int v = 0; v < V_; ++v) {
    float s = 0.f, sq = 0.f;
    for (int d = t; d < D_; d += 256) {
      float g = s_xin[v * D_ + d] * (2.0f * sigmoidf_(p.rn_gate[v * D_ + d]));
      s += g; sq += g * g;
    }
    float S = blk_sum(s, s_red);
    float Q = blk_sum(sq, s_red);
    if (t == 0) {
      float mu = S / 512.0f;
      s_mu[v] = mu; s_rs[v] = rsqrtf(Q / 512.0f - mu * mu + 1e-5f);
    }
  }
  __syncthreads();
  for (int j = 0; j < V_; ++j) {
    float pj = 0.f;
    for (int k = t; k < V_ * D_; k += 256) pj += s_xin[k] * p.fc1_w[j * (V_ * D_) + k];
    float S = blk_sum(pj, s_red);
    if (t == 0) {
      float x = S + p.fc1_b[j];
      s_h[j] = (x > 0.f) ? x : (__expf(x) - 1.0f);         // elu
    }
  }
  __syncthreads();
  if (t == 0) {
    float h2[V_], g[2 * V_];
    for (int j = 0; j < V_; ++j) {
      float a = p.fc2_b[j];
      for (int k = 0; k < V_; ++k) a += p.fc2_w[j * V_ + k] * s_h[k];
      h2[j] = a;
    }
    for (int j = 0; j < 2 * V_; ++j) {
      float a = p.glu_b[j];
      for (int k = 0; k < V_; ++k) a += p.glu_w[j * V_ + k] * h2[k];
      g[j] = a;
    }
    float res[V_]; float rmu = 0.f;
    const float scale = (float)(V_ * D_ - 1) / (float)(V_ - 1);
    for (int i = 0; i < V_; ++i) {
      float pos = i * scale;
      int lo = (int)floorf(pos);
      int hi = lo + 1; if (hi > V_ * D_ - 1) hi = V_ * D_ - 1;
      float fr = pos - (float)lo;
      float val = s_xin[lo] * (1.f - fr) + s_xin[hi] * fr;
      res[i] = val * (2.0f * sigmoidf_(p.rs_gate[i]));
      rmu += res[i];
    }
    rmu /= (float)V_;
    float rv = 0.f;
    for (int i = 0; i < V_; ++i) { float d0 = res[i] - rmu; rv += d0 * d0; }
    float rrs = rsqrtf(rv / (float)V_ + 1e-5f);
    float pre[V_]; float pmu = 0.f;
    for (int i = 0; i < V_; ++i) {
      float r = (res[i] - rmu) * rrs * p.rs_scale[i] + p.rs_bias[i];
      pre[i] = g[i] * sigmoidf_(g[V_ + i]) + r;
      pmu += pre[i];
    }
    pmu /= (float)V_;
    float pv = 0.f;
    for (int i = 0; i < V_; ++i) { float d0 = pre[i] - pmu; pv += d0 * d0; }
    float prs = rsqrtf(pv / (float)V_ + 1e-5f);
    float z[V_]; float zmax = -1e30f;
    for (int i = 0; i < V_; ++i) {
      z[i] = (pre[i] - pmu) * prs * p.grn_s[i] + p.grn_b[i];
      zmax = fmaxf(zmax, z[i]);
    }
    float zs = 0.f;
    for (int i = 0; i < V_; ++i) { z[i] = __expf(z[i] - zmax); zs += z[i]; }
    for (int i = 0; i < V_; ++i) s_w[i] = z[i] / zs;
  }
  __syncthreads();
  float o[2];
  for (int q = 0; q < 2; ++q) {
    int d = t + q * 256;
    float acc = 0.f;
    for (int v = 0; v < V_; ++v) {
      int idx = v * D_ + d;
      float gv = s_xin[idx] * (2.0f * sigmoidf_(p.rn_gate[idx]));
      float ln = (gv - s_mu[v]) * s_rs[v] * p.rn_scale[idx] + p.rn_bias[idx];
      acc += ln * s_w[v];
    }
    o[q] = acc;
  }
  __syncthreads();
  s_xin[t] = o[0]; s_xin[t + 256] = o[1];
  __syncthreads();
  float S = blk_sum(o[0] + o[1], s_red);
  float Q = blk_sum(o[0] * o[0] + o[1] * o[1], s_red);
  float mu = S / 512.f;
  float rsd = rsqrtf(Q / 512.f - mu * mu + 1e-5f);
  for (int q = 0; q < 2; ++q) {
    int d = t + q * 256;
    float f = (s_xin[d] - mu) * rsd * oln_s[d] + oln_b[d];
    out_bf16[d] = f2bf(f);
  }
  if (wgt_out && t < V_) wgt_out[t] = s_w[t];
}

__global__ void k_varsel0(const float* emb, const float* Xcov, const float* cov_w,
                          const float* Xs, VSN p, const float* ln1_s, const float* ln1_b,
                          unsigned short* Xcurb, float* wgt_base) {
  __shared__ float s_xin[V_ * D_];
  __shared__ float s_red[256];
  __shared__ float s_mu[V_], s_rs[V_], s_w[V_], s_h[V_];
  int bt = blockIdx.x;                                     // 0..383
  int b = bt / T_, tt = bt % T_;
  const float* lag = Xs + (size_t)(b * 18 + tt) * D_;
  for (int idx = threadIdx.x; idx < V_ * D_; idx += 256) {
    int v = idx >> 9, d = idx & 511;
    float val = (v == 0) ? emb[(size_t)bt * D_ + d]
                         : Xcov[(size_t)bt * I_ + (v - 1)] * cov_w[(v - 1) * D_ + d];
    s_xin[idx] = val + lag[d];
  }
  __syncthreads();
  float* wo = (tt == T_ - 1) ? (wgt_base + (size_t)(b * L_) * V_) : nullptr;
  varsel_core(s_xin, s_red, s_mu, s_rs, s_w, s_h, p, ln1_s, ln1_b,
              Xcurb + (size_t)bt * D_, wo);
}

__global__ void k_varsel_dec(int l, const float* yemb, const float* Xcov, const float* cov_w,
                             const float* Xs, VSN p, const float* ln3_s, const float* ln3_b,
                             unsigned short* decx, float* wgt_base) {
  __shared__ float s_xin[V_ * D_];
  __shared__ float s_red[256];
  __shared__ float s_mu[V_], s_rs[V_], s_w[V_], s_h[V_];
  int b = blockIdx.x;                                      // 0..31
  const float* lead = Xs + (size_t)(b * 18 + T_ + l) * D_;
  for (int idx = threadIdx.x; idx < V_ * D_; idx += 256) {
    int v = idx >> 9, d = idx & 511;
    float val = (v == 0) ? yemb[(size_t)(b * L_ + l) * D_ + d]
                         : Xcov[(size_t)(b * T_ + (T_ - 1)) * I_ + (v - 1)] * cov_w[(v - 1) * D_ + d];
    s_xin[idx] = val + lead[d];
  }
  __syncthreads();
  float* wo = wgt_base + (size_t)(b * L_ + l + 1) * V_;
  varsel_core(s_xin, s_red, s_mu, s_rs, s_w, s_h, p, ln3_s, ln3_b,
              decx + (size_t)b * D_, wo);
}

// ---------------------------------------------------------------- LSTM
__global__ void k_lstm_init(const unsigned short* xt, int xstride, float* c, unsigned short* xh) {
  int i = blockIdx.x * blockDim.x + threadIdx.x;
  if (i >= B_ * D_) return;
  int b = i >> 9, d = i & 511;
  c[i] = 0.f;
  xh[b * 1024 + 512 + d] = f2bf(0.f);
  xh[b * 1024 + d] = xt[(size_t)b * xstride + d];
}

// gates GEMM: M=32, N=2048, K=1024 over [x,h] bf16
__global__ void __launch_bounds__(256, 1)
k_lstm_gates(const unsigned short* xh, const unsigned short* WLb,
             const float* bih, const float* bhh, float* gates) {
  int wave = (blockIdx.x * blockDim.x + threadIdx.x) >> 5;
  int lane = threadIdx.x & 31;
  int tileM = wave & 1;
  int tileN = wave >> 1;
  if (tileN >= 128) return;
  const unsigned short* arow = xh + (size_t)(tileM * 16 + (lane & 15)) * 1024;
  const unsigned short* brow = WLb + (size_t)(tileN * 16 + (lane & 15)) * 1024;
  v8f acc = {0.f,0.f,0.f,0.f,0.f,0.f,0.f,0.f};
  for (int k = 0; k < 1024; k += 32) {
    acc = __builtin_amdgcn_wmma_f32_16x16x32_bf16(false, load_a_frag(arow, k, lane),
                                                  false, load_b_frag(brow, k, lane),
                                                  (short)0, acc, false, false);
  }
  int n = tileN * 16 + (lane & 15);
  float bb = bih[n] + bhh[n];
  for (int r = 0; r < 8; ++r) {
    int m = tileM * 16 + r + ((lane >> 4) << 3);
    gates[(size_t)m * 2048 + n] = acc[r] + bb;
  }
}

__global__ void k_lstm_update(const float* gates, float* c, float* h, unsigned short* xh,
                              const unsigned short* next_xt, int xstride) {
  int i = blockIdx.x * blockDim.x + threadIdx.x;
  if (i >= B_ * D_) return;
  int b = i >> 9, d = i & 511;
  const float* gr = gates + (size_t)b * 2048;
  float ig = sigmoidf_(gr[d]);
  float fg = sigmoidf_(gr[512 + d]);
  float gg = tanhf(gr[1024 + d]);
  float og = sigmoidf_(gr[1536 + d]);
  float cc = fg * c[i] + ig * gg;
  float hh = og * tanhf(cc);
  c[i] = cc; h[i] = hh;
  xh[b * 1024 + 512 + d] = f2bf(hh);
  if (next_xt) xh[b * 1024 + d] = next_xt[(size_t)b * xstride + d];
}

// ---------------------------------------------------------------- ydec = LN(h) + Xs_lead[l]
__global__ void k_ydec(int l, const float* h, const float* ln2_s, const float* ln2_b,
                       const float* Xs, float* yemb, unsigned short* yembb) {
  __shared__ float s_red[256];
  int b = blockIdx.x; int t = threadIdx.x;
  const float* hr = h + (size_t)b * D_;
  float v0 = hr[t], v1 = hr[t + 256];
  float S = blk_sum(v0 + v1, s_red);
  float Q = blk_sum(v0 * v0 + v1 * v1, s_red);
  float mu = S / 512.f, rs = rsqrtf(Q / 512.f - mu * mu + 1e-5f);
  const float* lead = Xs + (size_t)(b * 18 + T_ + l) * D_;
  size_t ro = (size_t)(b * L_ + l) * D_;
  for (int q = 0; q < 2; ++q) {
    int d = t + q * 256;
    float val = (hr[d] - mu) * rs * ln2_s[d] + ln2_b[d] + lead[d];
    yemb[ro + d] = val;
    yembb[ro + d] = f2bf(val);
  }
}

// ---------------------------------------------------------------- head GEMM: M=192, N=131072, K=512 -> d_out
// one wave = one N tile x ALL 12 M tiles -> head_w streamed exactly once.
// A (192x512 bf16) staged per-128-K-chunk into LDS with async-to-LDS (fallback: ds stores).
__global__ void __launch_bounds__(256, 1)
k_gemm_head(const unsigned short* Yb, const unsigned short* HWb,
            const float* head_b, float* out) {
  __shared__ unsigned short sA[192 * 128];                 // 48 KB
  int lane = threadIdx.x & 31;
  int wv = threadIdx.x >> 5;                               // 0..7
  int tileN = blockIdx.x * 8 + wv;                         // 0..8191
  const unsigned short* brow = HWb + (size_t)(tileN * 16 + (lane & 15)) * 512;
  v8f zer = {0.f,0.f,0.f,0.f,0.f,0.f,0.f,0.f};
  v8f acc[12];
#pragma unroll
  for (int m = 0; m < 12; ++m) acc[m] = zer;
  for (int kc = 0; kc < 512; kc += 128) {
    for (int it = threadIdx.x; it < 3072; it += 256) {     // 192 rows x 16 int4/row
      int row = it >> 4;
      int col = (it & 15) << 3;
      const unsigned short* g = Yb + (size_t)row * 512 + kc + col;
      unsigned short* l = sA + row * 128 + col;
#if HAVE_ASYNC_LDS
      async_copy16(g, l);
#else
      *(int4*)l = *(const int4*)g;
#endif
    }
#if HAVE_ASYNC_LDS
    async_wait0();
#endif
    __syncthreads();
#pragma unroll
    for (int k = 0; k < 128; k += 32) {
      v16bf b = load_b_frag(brow, kc + k, lane);
#pragma unroll
      for (int m = 0; m < 12; ++m) {
        const unsigned short* ar = sA + (m * 16 + (lane & 15)) * 128;
        v16bf a = load_a_frag(ar, k, lane);
        acc[m] = __builtin_amdgcn_wmma_f32_16x16x32_bf16(false, a, false, b, (short)0, acc[m], false, false);
      }
    }
    __syncthreads();
  }
  int n = tileN * 16 + (lane & 15);
  float bn = head_b[n];
#pragma unroll
  for (int m = 0; m < 12; ++m)
#pragma unroll
    for (int r = 0; r < 8; ++r)
      out[(size_t)(m * 16 + r + ((lane >> 4) << 3)) * 131072 + n] = acc[m][r] + bn;
}

// ---------------------------------------------------------------- host orchestration
extern "C" void kernel_launch(void* const* d_in, const int* in_sizes, int n_in,
                              void* d_out, int out_size, void* d_ws, size_t ws_size,
                              hipStream_t stream) {
  auto F = [&](int i) { return (const float*)d_in[i]; };
  char* ws = (char*)d_ws;
  size_t off = 0;
  auto alloc = [&](size_t bytes) -> char* {
    char* pp = ws + off; off = (off + bytes + 255) & ~(size_t)255; return pp;
  };
  unsigned short* A1   = (unsigned short*)alloc((size_t)196608 * 512 * 2);   // 201 MB im2col(h1*m1)
  unsigned short* A2   = (unsigned short*)alloc((size_t)384 * 131072 * 2);   // 100 MB h2*m2
  unsigned short* W3b  = (unsigned short*)alloc((size_t)512 * 131072 * 2);   // 134 MB
  unsigned short* HWb  = (unsigned short*)alloc((size_t)131072 * 512 * 2);   // 134 MB
  unsigned short* W2b  = (unsigned short*)alloc((size_t)256 * 512 * 2);
  unsigned short* WLb  = (unsigned short*)alloc((size_t)2048 * 1024 * 2);
  float* ratio1 = (float*)alloc(2048 * 4);
  float* upd1   = (float*)alloc(2048 * 4);
  float* ratio2 = (float*)alloc(512 * 4);
  float* upd2   = (float*)alloc(512 * 4);
  float* r3u3   = (float*)alloc(8);
  float* Xs     = (float*)alloc((size_t)B_ * 18 * D_ * 4);
  float* emb    = (float*)alloc((size_t)384 * 512 * 4);
  unsigned short* Xcurb = (unsigned short*)alloc((size_t)384 * 512 * 2);
  unsigned short* decx  = (unsigned short*)alloc((size_t)32 * 512 * 2);
  unsigned short* xh    = (unsigned short*)alloc((size_t)32 * 1024 * 2);
  float* cbuf  = (float*)alloc((size_t)32 * 512 * 4);
  float* hbuf  = (float*)alloc((size_t)32 * 512 * 4);
  float* gates = (float*)alloc((size_t)32 * 2048 * 4);
  float* yemb  = (float*)alloc((size_t)192 * 512 * 4);
  unsigned short* yembb = (unsigned short*)alloc((size_t)192 * 512 * 2);

  VSN p { F(12), F(13), F(14), F(15), F(16), F(17), F(18),
          F(19), F(20), F(21), F(22), F(23), F(24), F(25) };

  float* out = (float*)d_out;
  float* wgt_base = out + (size_t)192 * 131072;            // wgts (B,L,V) after Y

  // weight conversions to bf16 (vectorized)
  k_convert4<<<(16777216 + 255) / 256, 256, 0, stream>>>(F(36), HWb, (size_t)16777216); // head_w
  k_convert4<<<(16777216 + 255) / 256, 256, 0, stream>>>(F(8),  W3b, (size_t)16777216); // conv3_w
  k_convert4<<<(32768 + 255) / 256, 256, 0, stream>>>(F(6), W2b, (size_t)32768);        // conv2_w
  k_wcat<<<(2048 * 1024 + 255) / 256, 256, 0, stream>>>(F(32), F(33), WLb);

  // mask ratios, static embedding
  k_mask<<<1, 1024, 0, stream>>>(F(3), ratio1, upd1, ratio2, upd2, r3u3);
  k_xs<<<(B_ * 18 * D_ + 255) / 256, 256, 0, stream>>>(F(2), F(11), Xs);

  // conv stack
  k_h1<<<98304, 256, 0, stream>>>(F(0), F(3), F(4), F(5), ratio1, upd1, A1);
  k_gemm_conv2<<<3072, 256, 0, stream>>>(A1, W2b, F(7), ratio2, upd2, A2);              // 24576 waves
  k_zero<<<(196608 + 255) / 256, 256, 0, stream>>>(emb, (size_t)196608);
  k_gemm_conv3<<<384, 256, 0, stream>>>(A2, W3b, emb);                                  // 3072 waves
  k_finish3<<<(196608 + 255) / 256, 256, 0, stream>>>(emb, F(9), r3u3);

  // variable selection over history
  k_varsel0<<<384, 256, 0, stream>>>(emb, F(1), F(10), Xs, p, F(26), F(27), Xcurb, wgt_base);

  // LSTM over T=12 steps (l = 0)
  k_lstm_init<<<64, 256, 0, stream>>>(Xcurb, T_ * D_, cbuf, xh);
  for (int t = 0; t < T_; ++t) {
    k_lstm_gates<<<32, 256, 0, stream>>>(xh, WLb, F(34), F(35), gates);
    const unsigned short* nxt = (t < T_ - 1) ? (Xcurb + (size_t)(t + 1) * D_) : nullptr;
    k_lstm_update<<<64, 256, 0, stream>>>(gates, cbuf, hbuf, xh, nxt, T_ * D_);
  }

  // decode loop
  for (int l = 0; l < L_; ++l) {
    k_ydec<<<32, 256, 0, stream>>>(l, hbuf, F(28), F(29), Xs, yemb, yembb);
    if (l < L_ - 1) {
      k_varsel_dec<<<32, 256, 0, stream>>>(l, yemb, F(1), F(10), Xs, p, F(30), F(31), decx, wgt_base);
      k_lstm_init<<<64, 256, 0, stream>>>(decx, D_, cbuf, xh);
      k_lstm_gates<<<32, 256, 0, stream>>>(xh, WLb, F(34), F(35), gates);
      k_lstm_update<<<64, 256, 0, stream>>>(gates, cbuf, hbuf, xh, nullptr, 0);
    }
  }

  // head projection straight into d_out (head_w streamed exactly once)
  k_gemm_head<<<1024, 256, 0, stream>>>(yembb, HWb, F(37), out);                        // 8192 waves
}
